// OverlapGATNet_33200097198390
// MI455X (gfx1250) — matrix operand order repaired and checked by reading, not compiled, not run on hardware
//
#include <hip/hip_runtime.h>

// ---------------------------------------------------------------------------
// Types for CDNA5 WMMA (wave32)
// ---------------------------------------------------------------------------
typedef __attribute__((ext_vector_type(16))) __bf16    v16bf;
typedef __attribute__((ext_vector_type(8)))  float     v8f;
typedef __attribute__((ext_vector_type(4)))  unsigned  v4u;

union Frag {
    v16bf bf;
    v4u   u[2];
};

// ---------------------------------------------------------------------------
// Helpers
// ---------------------------------------------------------------------------
__device__ inline unsigned short f32_to_bf16(float f) {
    unsigned u = __float_as_uint(f);
    unsigned r = u + 0x7FFFu + ((u >> 16) & 1u);   // round-to-nearest-even
    return (unsigned short)(r >> 16);
}

// monotone float <-> uint key mapping (for atomicMax on signed floats)
__device__ inline unsigned floatToKey(float f) {
    unsigned u = __float_as_uint(f);
    return (u & 0x80000000u) ? ~u : (u | 0x80000000u);
}
__device__ inline float keyToFloat(unsigned k) {
    unsigned u = (k & 0x80000000u) ? (k & 0x7FFFFFFFu) : ~k;
    return __uint_as_float(u);
}

// ---------------------------------------------------------------------------
// fp32 -> bf16 elementwise convert
// ---------------------------------------------------------------------------
__global__ __launch_bounds__(256) void cvt_bf16_kernel(
    const float* __restrict__ in, unsigned short* __restrict__ out, int n)
{
    int i = blockIdx.x * blockDim.x + threadIdx.x;
    if (i < n) out[i] = f32_to_bf16(in[i]);
}

// ---------------------------------------------------------------------------
// Pack weight W[K x M] (row major fp32) into WMMA B-fragment layout (bf16):
// packed[((ntile*(K/32)+ks)*32 + lane)*16 + h] = W[ks*32 + (lane>>4)*16 + h][ntile*16 + (lane&15)]
// ---------------------------------------------------------------------------
__global__ __launch_bounds__(256) void pack_w_kernel(
    const float* __restrict__ W, unsigned short* __restrict__ out, int K, int M)
{
    int idx = blockIdx.x * blockDim.x + threadIdx.x;
    if (idx >= K * M) return;
    int h    = idx & 15;
    int lane = (idx >> 4) & 31;
    int rest = idx >> 9;
    int nks  = K / 32;
    int ks   = rest % nks;
    int nt   = rest / nks;
    int col  = nt * 16 + (lane & 15);
    int k    = ks * 32 + ((lane >> 4) << 4) + h;
    out[idx] = f32_to_bf16(W[(size_t)k * M + col]);
}

// ---------------------------------------------------------------------------
// Fused triple GEMM:  xl = X@Wl, xr = X@Wr, outres = X@Wres + bias
// X in bf16 row-major [N x K]; weights pre-packed into B-fragment layout.
// One wave per 16x16 output tile; v_wmma_f32_16x16x32_bf16, K/32 steps.
// ---------------------------------------------------------------------------
template<int K, int M>
__global__ __launch_bounds__(256) void gemm3_kernel(
    const unsigned short* __restrict__ Xb,
    const unsigned short* __restrict__ Wlp,
    const unsigned short* __restrict__ Wrp,
    const unsigned short* __restrict__ Wsp,
    const float* __restrict__ bias,
    float* __restrict__ xl, float* __restrict__ xr, float* __restrict__ outres,
    int nTilesM)
{
    const int lane  = threadIdx.x & 31;
    const int wid   = (blockIdx.x * blockDim.x + threadIdx.x) >> 5;
    const int ntN   = M / 16;
    const int mtile = wid / ntN;
    const int ntile = wid % ntN;
    if (mtile >= nTilesM) return;

    const int khalf = lane >> 4;
    const int nlo   = lane & 15;
    const int row   = mtile * 16 + nlo;

    constexpr int NKS = K / 32;
    Frag A[NKS];
    const unsigned short* xp = Xb + (size_t)row * K;
#pragma unroll
    for (int ks = 0; ks < NKS; ks++) {
        const unsigned short* p = xp + ks * 32 + khalf * 16;
        A[ks].u[0] = *(const v4u*)(p);        // halves K = base+0..7
        A[ks].u[1] = *(const v4u*)(p + 8);    // halves K = base+8..15
    }

    v8f accL = {}, accR = {}, accS = {};
#pragma unroll
    for (int ks = 0; ks < NKS; ks++) {
        size_t boff = ((size_t)(ntile * NKS + ks) * 32 + lane) * 16;
        Frag Bl, Br, Bs;
        Bl.u[0] = *(const v4u*)(Wlp + boff); Bl.u[1] = *(const v4u*)(Wlp + boff + 8);
        Br.u[0] = *(const v4u*)(Wrp + boff); Br.u[1] = *(const v4u*)(Wrp + boff + 8);
        Bs.u[0] = *(const v4u*)(Wsp + boff); Bs.u[1] = *(const v4u*)(Wsp + boff + 8);
        accL = __builtin_amdgcn_wmma_f32_16x16x32_bf16(false, A[ks].bf, false, Bl.bf,
                                                       (short)0, accL, false, false);
        accR = __builtin_amdgcn_wmma_f32_16x16x32_bf16(false, A[ks].bf, false, Br.bf,
                                                       (short)0, accR, false, false);
        accS = __builtin_amdgcn_wmma_f32_16x16x32_bf16(false, A[ks].bf, false, Bs.bf,
                                                       (short)0, accS, false, false);
    }

    const int col = ntile * 16 + nlo;
    const float b = bias[col];
#pragma unroll
    for (int v = 0; v < 8; v++) {
        size_t r = (size_t)(mtile * 16 + v + 8 * khalf) * M + col;
        xl[r]     = accL[v];
        xr[r]     = accR[v];
        outres[r] = accS[v] + b;
    }
}

// ---------------------------------------------------------------------------
// Segment-softmax init
// ---------------------------------------------------------------------------
__global__ __launch_bounds__(256) void init_kernel(
    unsigned* __restrict__ mkey, float* __restrict__ denom, int n)
{
    int i = blockIdx.x * blockDim.x + threadIdx.x;
    if (i < n) { mkey[i] = 0u; denom[i] = 0.0f; }
}

// ---------------------------------------------------------------------------
// Edge score: e = att . LeakyReLU(xl[src] + xr[dst]); atomic segment-max.
// 8 lanes per edge, float4 loads.
// ---------------------------------------------------------------------------
template<int F>
__global__ __launch_bounds__(256) void edge_score_kernel(
    const float* __restrict__ xl, const float* __restrict__ xr,
    const int* __restrict__ src, const int* __restrict__ dst,
    const float* __restrict__ att, float* __restrict__ esc,
    unsigned* __restrict__ mkey, int nE)
{
    int t = blockIdx.x * blockDim.x + threadIdx.x;
    int e = t >> 3, sub = t & 7;
    if (e >= nE) return;
    int s = src[e], d = dst[e];
    const float* pl = xl + (size_t)s * F;
    const float* pr = xr + (size_t)d * F;
    float acc = 0.0f;
#pragma unroll
    for (int k = 0; k < F / 8; k += 4) {
        int kk = sub * (F / 8) + k;
        float4 a = *(const float4*)(pl + kk);
        float4 b = *(const float4*)(pr + kk);
        float4 w = *(const float4*)(att + kk);
        float h;
        h = a.x + b.x; acc += w.x * (h > 0.0f ? h : 0.2f * h);
        h = a.y + b.y; acc += w.y * (h > 0.0f ? h : 0.2f * h);
        h = a.z + b.z; acc += w.z * (h > 0.0f ? h : 0.2f * h);
        h = a.w + b.w; acc += w.w * (h > 0.0f ? h : 0.2f * h);
    }
    acc += __shfl_down(acc, 4, 8);
    acc += __shfl_down(acc, 2, 8);
    acc += __shfl_down(acc, 1, 8);
    if (sub == 0) {
        esc[e] = acc;
        atomicMax(&mkey[d], floatToKey(acc));
    }
}

// ---------------------------------------------------------------------------
// Softmax numerator + segment denominator
// ---------------------------------------------------------------------------
__global__ __launch_bounds__(256) void edge_norm_kernel(
    float* __restrict__ esc, const int* __restrict__ dst,
    const unsigned* __restrict__ mkey, float* __restrict__ denom, int nE)
{
    int e = blockIdx.x * blockDim.x + threadIdx.x;
    if (e >= nE) return;
    int d = dst[e];
    float m = keyToFloat(mkey[d]);
    float w = __expf(esc[e] - m);
    esc[e] = w;
    atomicAdd(&denom[d], w);
}

// ---------------------------------------------------------------------------
// Weighted scatter-add:  out[dst] += alpha * xl[src]
// ---------------------------------------------------------------------------
template<int F>
__global__ __launch_bounds__(256) void edge_agg_kernel(
    const float* __restrict__ xl, const int* __restrict__ src,
    const int* __restrict__ dst, const float* __restrict__ esc,
    const float* __restrict__ denom, float* __restrict__ out, int nE)
{
    int t = blockIdx.x * blockDim.x + threadIdx.x;
    int e = t >> 3, sub = t & 7;
    if (e >= nE) return;
    int s = src[e], d = dst[e];
    float alpha = esc[e] / (denom[d] + 1e-16f);
    const float* pl = xl + (size_t)s * F;
    float* po = out + (size_t)d * F;
#pragma unroll
    for (int k = 0; k < F / 8; k += 4) {
        int kk = sub * (F / 8) + k;
        float4 a = *(const float4*)(pl + kk);
        atomicAdd(po + kk + 0, alpha * a.x);
        atomicAdd(po + kk + 1, alpha * a.y);
        atomicAdd(po + kk + 2, alpha * a.z);
        atomicAdd(po + kk + 3, alpha * a.w);
    }
}

// ---------------------------------------------------------------------------
// ELU + convert to bf16 (between layers)
// ---------------------------------------------------------------------------
__global__ __launch_bounds__(256) void elu_cvt_kernel(
    const float* __restrict__ in, unsigned short* __restrict__ out, int n)
{
    int i = blockIdx.x * blockDim.x + threadIdx.x;
    if (i >= n) return;
    float v = in[i];
    float r = v > 0.0f ? v : (__expf(v) - 1.0f);
    out[i] = f32_to_bf16(r);
}

// ---------------------------------------------------------------------------
// Host launcher
// ---------------------------------------------------------------------------
extern "C" void kernel_launch(void* const* d_in, const int* in_sizes, int n_in,
                              void* d_out, int out_size, void* d_ws, size_t ws_size,
                              hipStream_t stream)
{
    constexpr int N  = 50000;
    constexpr int E  = 640000;
    constexpr int DI = 128;
    constexpr int H1 = 128;
    constexpr int H2 = 256;

    const float* x     = (const float*)d_in[0];
    const int*   ei    = (const int*)d_in[1];
    const int*   src   = ei;
    const int*   dst   = ei + E;
    const float* Wl1   = (const float*)d_in[2];
    const float* Wr1   = (const float*)d_in[3];
    const float* att1  = (const float*)d_in[4];
    const float* Wres1 = (const float*)d_in[5];
    const float* b1    = (const float*)d_in[6];
    const float* Wl2   = (const float*)d_in[7];
    const float* Wr2   = (const float*)d_in[8];
    const float* att2  = (const float*)d_in[9];
    const float* Wres2 = (const float*)d_in[10];
    const float* b2    = (const float*)d_in[11];
    float* out = (float*)d_out;

    // workspace carve (256B aligned)
    char* base = (char*)d_ws;
    size_t off = 0;
    auto carve = [&](size_t bytes) -> void* {
        void* r = base + off;
        off = (off + bytes + 255) & ~(size_t)255;
        return r;
    };
    unsigned short* xbf  = (unsigned short*)carve((size_t)N * DI * 2);
    unsigned short* hbf  = (unsigned short*)carve((size_t)N * H1 * 2);
    unsigned short* Wl1p = (unsigned short*)carve((size_t)DI * H1 * 2);
    unsigned short* Wr1p = (unsigned short*)carve((size_t)DI * H1 * 2);
    unsigned short* Ws1p = (unsigned short*)carve((size_t)DI * H1 * 2);
    unsigned short* Wl2p = (unsigned short*)carve((size_t)H1 * H2 * 2);
    unsigned short* Wr2p = (unsigned short*)carve((size_t)H1 * H2 * 2);
    unsigned short* Ws2p = (unsigned short*)carve((size_t)H1 * H2 * 2);
    float* xl1   = (float*)carve((size_t)N * H1 * 4);
    float* xr1   = (float*)carve((size_t)N * H1 * 4);
    float* out1  = (float*)carve((size_t)N * H1 * 4);
    float* xl2   = (float*)carve((size_t)N * H2 * 4);
    float* xr2   = (float*)carve((size_t)N * H2 * 4);
    float* esc   = (float*)carve((size_t)E * 4);
    unsigned* mkey = (unsigned*)carve((size_t)N * 4);
    float* denom = (float*)carve((size_t)N * 4);

    const int TB = 256;

    // prep: bf16 convert + weight packing
    cvt_bf16_kernel<<<(N * DI + TB - 1) / TB, TB, 0, stream>>>(x, xbf, N * DI);
    pack_w_kernel<<<(DI * H1 + TB - 1) / TB, TB, 0, stream>>>(Wl1,   Wl1p, DI, H1);
    pack_w_kernel<<<(DI * H1 + TB - 1) / TB, TB, 0, stream>>>(Wr1,   Wr1p, DI, H1);
    pack_w_kernel<<<(DI * H1 + TB - 1) / TB, TB, 0, stream>>>(Wres1, Ws1p, DI, H1);
    pack_w_kernel<<<(H1 * H2 + TB - 1) / TB, TB, 0, stream>>>(Wl2,   Wl2p, H1, H2);
    pack_w_kernel<<<(H1 * H2 + TB - 1) / TB, TB, 0, stream>>>(Wr2,   Wr2p, H1, H2);
    pack_w_kernel<<<(H1 * H2 + TB - 1) / TB, TB, 0, stream>>>(Wres2, Ws2p, H1, H2);

    // ---- Layer 1 ----
    {
        int tiles = (N / 16) * (H1 / 16);                    // 25000 -> 3125 blocks
        gemm3_kernel<DI, H1><<<tiles / 8, TB, 0, stream>>>(
            xbf, Wl1p, Wr1p, Ws1p, b1, xl1, xr1, out1, N / 16);
        init_kernel<<<(N + TB - 1) / TB, TB, 0, stream>>>(mkey, denom, N);
        edge_score_kernel<H1><<<(E * 8) / TB, TB, 0, stream>>>(
            xl1, xr1, src, dst, att1, esc, mkey, E);
        edge_norm_kernel<<<(E + TB - 1) / TB, TB, 0, stream>>>(esc, dst, mkey, denom, E);
        edge_agg_kernel<H1><<<(E * 8) / TB, TB, 0, stream>>>(
            xl1, src, dst, esc, denom, out1, E);
        elu_cvt_kernel<<<(N * H1 + TB - 1) / TB, TB, 0, stream>>>(out1, hbf, N * H1);
    }

    // ---- Layer 2 ----
    {
        int tiles = (N / 16) * (H2 / 16);                    // 50000 -> 6250 blocks
        gemm3_kernel<H1, H2><<<tiles / 8, TB, 0, stream>>>(
            hbf, Wl2p, Wr2p, Ws2p, b2, xl2, xr2, out, N / 16);
        init_kernel<<<(N + TB - 1) / TB, TB, 0, stream>>>(mkey, denom, N);
        edge_score_kernel<H2><<<(E * 8) / TB, TB, 0, stream>>>(
            xl2, xr2, src, dst, att2, esc, mkey, E);
        edge_norm_kernel<<<(E + TB - 1) / TB, TB, 0, stream>>>(esc, dst, mkey, denom, E);
        edge_agg_kernel<H2><<<(E * 8) / TB, TB, 0, stream>>>(
            xl2, src, dst, esc, denom, out, E);
    }
}
